// GINwithClassifier_77137612636243
// MI455X (gfx1250) — matrix operand order repaired and checked by reading, not compiled, not run on hardware
//
#include <hip/hip_runtime.h>
#include <hip/hip_bf16.h>

// ---------------- problem constants ----------------
#define N_NODES   50000
#define N_EDGES   256000
#define N_GRAPHS  256
#define IN_DIM    128
#define HID       300
#define HP        320          // HID padded to multiple of 32 (WMMA K-step)
#define MP        50048        // N_NODES padded to multiple of 64 (block M tile)
#define N_LAYERS  5
#define N_CLASSES 10
#define BN_EPS    1e-5f

typedef __attribute__((ext_vector_type(16))) __bf16 v16bf;
typedef __attribute__((ext_vector_type(8)))  __bf16 v8bf;
typedef __attribute__((ext_vector_type(8)))  float  v8f;

// ---------------- elementwise / packing kernels ----------------

// x [N_NODES, IN_DIM] fp32 -> h [MP, HP] bf16, zero padded (rows >= N, cols >= IN_DIM)
__global__ __launch_bounds__(256) void k_cast_x(const float* __restrict__ x,
                                                __bf16* __restrict__ h) {
  int t = blockIdx.x * blockDim.x + threadIdx.x;
  if (t >= MP * HP) return;
  int row = t / HP, col = t - row * HP;
  float v = (row < N_NODES && col < IN_DIM) ? x[(size_t)row * IN_DIM + col] : 0.f;
  h[t] = (__bf16)v;
}

// W [din, dout] fp32 row-major -> Wt [HP, Kp] bf16 (transposed, zero padded)
__global__ __launch_bounds__(256) void k_pack_w(const float* __restrict__ W,
                                                __bf16* __restrict__ Wt,
                                                int din, int dout, int Kp) {
  int t = blockIdx.x * blockDim.x + threadIdx.x;
  if (t >= HP * Kp) return;
  int n = t / Kp, k = t - n * Kp;
  float v = (n < dout && k < din) ? W[(size_t)k * dout + n] : 0.f;
  Wt[t] = (__bf16)v;
}

__global__ __launch_bounds__(256) void k_pack_bias(const float* __restrict__ b,
                                                   float* __restrict__ bb, int n) {
  int t = blockIdx.x * blockDim.x + threadIdx.x;
  if (t >= HP) return;
  bb[t] = (t < n) ? b[t] : 0.f;
}

// fold eval-mode BN into scale/shift applied after the Wc1 GEMM:
// z = relu((acc + bc1 - mean) * gamma*rsqrt(var+eps) + beta) = relu(acc*scale + shift)
__global__ __launch_bounds__(256) void k_bn_fold(const float* __restrict__ bc1,
                                                 const float* __restrict__ gamma,
                                                 const float* __restrict__ beta,
                                                 const float* __restrict__ mean,
                                                 const float* __restrict__ var,
                                                 float* __restrict__ scale,
                                                 float* __restrict__ shift) {
  int t = blockIdx.x * blockDim.x + threadIdx.x;
  if (t >= HP) return;
  if (t < HID) {
    float s = gamma[t] * rsqrtf(var[t] + BN_EPS);
    scale[t] = s;
    shift[t] = (bc1[t] - mean[t]) * s + beta[t];
  } else {
    scale[t] = 0.f;
    shift[t] = 0.f;
  }
}

// ---------------- edge scatter-add: agg[dst] += h[src] ----------------
// agg was pre-copied from h, so after this agg == (1+eps)*h + sum_neigh (eps=0).
// One thread per (edge, bf16-pair); packed bf16 atomic add (CDNA5).
__global__ __launch_bounds__(256) void k_scatter(const int* __restrict__ ei,
                                                 const __bf16* __restrict__ h,
                                                 __bf16* __restrict__ agg,
                                                 int pairs) {
  long long t = (long long)blockIdx.x * blockDim.x + threadIdx.x;
  if (t >= (long long)N_EDGES * pairs) return;
  int e = (int)(t / pairs);
  int p = (int)(t - (long long)e * pairs);
  int s = ei[e];                // edge_index[0][e]
  int d = ei[N_EDGES + e];      // edge_index[1][e]
  unsigned val = *(const unsigned*)(h + (size_t)s * HP + 2 * p);
  const __bf16* a = agg + (size_t)d * HP + 2 * p;
  asm volatile("global_atomic_pk_add_bf16 %0, %1, off"
               :: "v"(a), "v"(val) : "memory");
}

// ---------------- WMMA bf16 GEMM (fused scale/shift + ReLU) ----------------
// A-frag layout (16x32 bf16): lanes 0-15 row=lane, K = k0+0..7 / k0+16..23;
// lanes 16-31 row=lane-16, K = k0+8..15 / k0+24..31.
__device__ __forceinline__ v16bf load_a(const __bf16* __restrict__ A, int lda,
                                        int row0, int k0, int lane) {
  int half = lane >> 4, m = lane & 15;
  const __bf16* p = A + (size_t)(row0 + m) * lda + k0 + half * 8;
  v8bf lo = *(const v8bf*)p;
  v8bf hi = *(const v8bf*)(p + 16);
  v16bf r;
#pragma unroll
  for (int i = 0; i < 8; ++i) { r[i] = lo[i]; r[i + 8] = hi[i]; }
  return r;
}

// B-frag layout (32x16 bf16): lanes 0-15 col=lane, K = k0+0..7 / k0+16..23; lanes 16-31 K = k0+8..15 / k0+24..31.
// Wt is pre-transposed [N][Kp], so this is two contiguous 16B loads per lane.
__device__ __forceinline__ v16bf load_b(const __bf16* __restrict__ Wt, int ldw,
                                        int col0, int k0, int lane) {
  int half = lane >> 4, n = lane & 15;
  const __bf16* p = Wt + (size_t)(col0 + n) * ldw + k0 + half * 8;
  v8bf lo = *(const v8bf*)p;
  v8bf hi = *(const v8bf*)(p + 16);
  v16bf r;
#pragma unroll
  for (int i = 0; i < 8; ++i) { r[i] = lo[i]; r[i + 8] = hi[i]; }
  return r;
}

// C/D layout: VGPR r -> lanes 0-15 (M=r, N=lane); lanes 16-31 (M=8+r, N=lane-16)
__device__ __forceinline__ void store_tile(__bf16* __restrict__ Out, int ldo,
                                           int row0, int col0, v8f acc,
                                           const float* __restrict__ scale,
                                           const float* __restrict__ shift,
                                           int lane) {
  int half = lane >> 4, n = lane & 15;
  int col = col0 + n;
  float sc = scale ? scale[col] : 1.0f;
  float sh = shift[col];
#pragma unroll
  for (int r = 0; r < 8; ++r) {
    int row = row0 + half * 8 + r;
    float v = acc[r] * sc + sh;
    v = v > 0.f ? v : 0.f;  // ReLU (all fused epilogues in this model end in ReLU)
    Out[(size_t)row * ldo + col] = (__bf16)v;
  }
}

// Out[M,HP] = relu(scale * (A[M,K] @ Wt[HP,K]^T) + shift), bf16 in/out, fp32 accum.
// 128 threads = 4 waves; wave tile 32x32 = 2x2 WMMA; block tile 64x64.
__global__ __launch_bounds__(128) void k_gemm(const __bf16* __restrict__ A,
                                              const __bf16* __restrict__ Wt,
                                              const float* __restrict__ scale,
                                              const float* __restrict__ shift,
                                              __bf16* __restrict__ Out,
                                              int lda, int ldw, int K) {
  int wave = threadIdx.x >> 5, lane = threadIdx.x & 31;
  int row0 = blockIdx.y * 64 + (wave >> 1) * 32;
  int col0 = blockIdx.x * 64 + (wave & 1) * 32;
  v8f acc00 = {}, acc01 = {}, acc10 = {}, acc11 = {};
  for (int k = 0; k < K; k += 32) {
    v16bf a0 = load_a(A, lda, row0, k, lane);
    v16bf a1 = load_a(A, lda, row0 + 16, k, lane);
    v16bf b0 = load_b(Wt, ldw, col0, k, lane);
    v16bf b1 = load_b(Wt, ldw, col0 + 16, k, lane);
    acc00 = __builtin_amdgcn_wmma_f32_16x16x32_bf16(false, a0, false, b0, (short)0, acc00, false, false);
    acc01 = __builtin_amdgcn_wmma_f32_16x16x32_bf16(false, a0, false, b1, (short)0, acc01, false, false);
    acc10 = __builtin_amdgcn_wmma_f32_16x16x32_bf16(false, a1, false, b0, (short)0, acc10, false, false);
    acc11 = __builtin_amdgcn_wmma_f32_16x16x32_bf16(false, a1, false, b1, (short)0, acc11, false, false);
  }
  store_tile(Out, HP, row0,      col0,      acc00, scale, shift, lane);
  store_tile(Out, HP, row0,      col0 + 16, acc01, scale, shift, lane);
  store_tile(Out, HP, row0 + 16, col0,      acc10, scale, shift, lane);
  store_tile(Out, HP, row0 + 16, col0 + 16, acc11, scale, shift, lane);
}

// ---------------- pooling + classifier tail ----------------
__global__ __launch_bounds__(256) void k_zero_f32(float* __restrict__ p, int n) {
  int t = blockIdx.x * blockDim.x + threadIdx.x;
  if (t < n) p[t] = 0.f;
}

__global__ __launch_bounds__(256) void k_pool(const __bf16* __restrict__ h,
                                              const int* __restrict__ batch,
                                              float* __restrict__ g) {
  long long t = (long long)blockIdx.x * blockDim.x + threadIdx.x;
  if (t >= (long long)N_NODES * HID) return;
  int node = (int)(t / HID);
  int d = (int)(t - (long long)node * HID);
  atomicAdd(&g[(size_t)batch[node] * HP + d], (float)h[(size_t)node * HP + d]);
}

__global__ __launch_bounds__(256) void k_cast_g(const float* __restrict__ g,
                                                __bf16* __restrict__ gb) {
  int t = blockIdx.x * blockDim.x + threadIdx.x;
  if (t >= N_GRAPHS * HP) return;
  gb[t] = (__bf16)g[t];
}

// out[G, 10] = z[G, HID] @ Wc2[HID, 10] + bc2   (tiny, fp32)
__global__ __launch_bounds__(256) void k_final(const __bf16* __restrict__ z,
                                               const float* __restrict__ Wc2,
                                               const float* __restrict__ bc2,
                                               float* __restrict__ out) {
  int t = blockIdx.x * blockDim.x + threadIdx.x;
  if (t >= N_GRAPHS * N_CLASSES) return;
  int gi = t / N_CLASSES, c = t - gi * N_CLASSES;
  float s = bc2[c];
  for (int k = 0; k < HID; ++k)
    s += (float)z[(size_t)gi * HP + k] * Wc2[(size_t)k * N_CLASSES + c];
  out[t] = s;
}

// ---------------- launch ----------------
static inline int blks(long long n, int b) { return (int)((n + b - 1) / b); }

extern "C" void kernel_launch(void* const* d_in, const int* in_sizes, int n_in,
                              void* d_out, int out_size, void* d_ws, size_t ws_size,
                              hipStream_t stream) {
  const float* x     = (const float*)d_in[0];
  const int*   ei    = (const int*)d_in[1];
  const int*   batch = (const int*)d_in[2];
  const float *W1[N_LAYERS], *b1[N_LAYERS], *W2[N_LAYERS], *b2[N_LAYERS];
  int idx = 3;
  for (int l = 0; l < N_LAYERS; ++l) {
    W1[l] = (const float*)d_in[idx++]; b1[l] = (const float*)d_in[idx++];
    W2[l] = (const float*)d_in[idx++]; b2[l] = (const float*)d_in[idx++];
  }
  const float* Wc1   = (const float*)d_in[idx++];
  const float* bc1   = (const float*)d_in[idx++];
  const float* gamma = (const float*)d_in[idx++];
  const float* beta  = (const float*)d_in[idx++];
  const float* mean  = (const float*)d_in[idx++];
  const float* var   = (const float*)d_in[idx++];
  const float* Wc2   = (const float*)d_in[idx++];
  const float* bc2   = (const float*)d_in[idx++];

  // workspace carve-out
  char* ws = (char*)d_ws;
  size_t off = 0;
  auto take = [&](size_t bytes) -> char* {
    char* p = ws + off;
    off = (off + bytes + 255) & ~(size_t)255;
    return p;
  };
  const size_t actBytes = (size_t)MP * HP * sizeof(__bf16);
  __bf16* hbuf = (__bf16*)take(actBytes);   // current node features
  __bf16* abuf = (__bf16*)take(actBytes);   // h + agg (GEMM1 input)
  __bf16* tbuf = (__bf16*)take(actBytes);   // relu(mW1+b1)
  __bf16* wpk[11];
  for (int i = 0; i < 11; ++i) wpk[i] = (__bf16*)take((size_t)HP * HP * sizeof(__bf16));
  float* bpk[12];
  for (int i = 0; i < 12; ++i) bpk[i] = (float*)take(HP * sizeof(float));
  float*  gbuf = (float*)take((size_t)N_GRAPHS * HP * sizeof(float));
  __bf16* gbf  = (__bf16*)take((size_t)N_GRAPHS * HP * sizeof(__bf16));
  __bf16* zbuf = (__bf16*)take((size_t)N_GRAPHS * HP * sizeof(__bf16));

  // pack weights / biases (cheap; done every call for determinism)
  for (int l = 0; l < N_LAYERS; ++l) {
    int din = l ? HID : IN_DIM;
    int Kp  = l ? HP : IN_DIM;
    k_pack_w<<<blks(HP * Kp, 256), 256, 0, stream>>>(W1[l], wpk[2 * l], din, HID, Kp);
    k_pack_w<<<blks(HP * HP, 256), 256, 0, stream>>>(W2[l], wpk[2 * l + 1], HID, HID, HP);
    k_pack_bias<<<blks(HP, 256), 256, 0, stream>>>(b1[l], bpk[2 * l], HID);
    k_pack_bias<<<blks(HP, 256), 256, 0, stream>>>(b2[l], bpk[2 * l + 1], HID);
  }
  k_pack_w<<<blks(HP * HP, 256), 256, 0, stream>>>(Wc1, wpk[10], HID, HID, HP);
  k_bn_fold<<<blks(HP, 256), 256, 0, stream>>>(bc1, gamma, beta, mean, var,
                                               bpk[11] /*scale*/, bpk[10] /*shift*/);

  // h0 = x (bf16, padded)
  k_cast_x<<<blks((long long)MP * HP, 256), 256, 0, stream>>>(x, hbuf);

  dim3 gemmGrid(HP / 64, MP / 64);
  for (int l = 0; l < N_LAYERS; ++l) {
    int din = l ? HID : IN_DIM;
    int Kp  = l ? HP : IN_DIM;
    // abuf = h; abuf[dst] += h[src] over edges
    hipMemcpyAsync(abuf, hbuf, actBytes, hipMemcpyDeviceToDevice, stream);
    int pairs = din / 2;
    k_scatter<<<blks((long long)N_EDGES * pairs, 256), 256, 0, stream>>>(ei, hbuf, abuf, pairs);
    // t = relu(abuf @ W1 + b1) ; h = relu(t @ W2 + b2)
    k_gemm<<<gemmGrid, 128, 0, stream>>>(abuf, wpk[2 * l], nullptr, bpk[2 * l], tbuf, HP, Kp, Kp);
    k_gemm<<<gemmGrid, 128, 0, stream>>>(tbuf, wpk[2 * l + 1], nullptr, bpk[2 * l + 1], hbuf, HP, HP, HP);
  }

  // global_add_pool + classifier
  k_zero_f32<<<blks(N_GRAPHS * HP, 256), 256, 0, stream>>>(gbuf, N_GRAPHS * HP);
  k_pool<<<blks((long long)N_NODES * HID, 256), 256, 0, stream>>>(hbuf, batch, gbuf);
  k_cast_g<<<blks(N_GRAPHS * HP, 256), 256, 0, stream>>>(gbuf, gbf);
  dim3 clsGrid(HP / 64, N_GRAPHS / 64);
  k_gemm<<<clsGrid, 128, 0, stream>>>(gbf, wpk[10], bpk[11], bpk[10], zbuf, HP, HP, HP);
  k_final<<<blks(N_GRAPHS * N_CLASSES, 256), 256, 0, stream>>>(zbuf, Wc2, bc2, (float*)d_out);
}
